// TemporalFusionLayer_22368189677643
// MI455X (gfx1250) — compile-verified, matrix operand order checked by reading, not run on hardware
//
#include <hip/hip_runtime.h>
#include <hip/hip_bf16.h>
#include <math.h>

#define N_NODES   262144
#define NC_NODES  65536
#define DIM       256
#define KTOT      768          // 3*DIM total K after fold
#define NKT       24           // KTOT/32 k-tiles
#define NNT       16           // DIM/16 n-tiles

typedef __attribute__((ext_vector_type(16))) __bf16 v16bf;
typedef __attribute__((ext_vector_type(8)))  float  v8f;
typedef __attribute__((ext_vector_type(4)))  float  v4f;   // native vector for NT builtins

__device__ __forceinline__ unsigned short f32_to_bf16_rne(float f) {
    unsigned int u = __float_as_uint(f);
    u += 0x7FFFu + ((u >> 16) & 1u);        // round to nearest even
    return (unsigned short)(u >> 16);
}

// ---------------------------------------------------------------------------
// Kernel 1: Wt[s][j][n] = W_fuse[n][s*DIM + j]   (coalesced writes, n fastest)
// ---------------------------------------------------------------------------
__global__ void k_transpose_wfuse(const float* __restrict__ Wf,
                                  float* __restrict__ Wt) {
    int t = blockIdx.x * blockDim.x + threadIdx.x;
    if (t >= 3 * DIM * DIM) return;
    int n = t & (DIM - 1);
    int j = (t >> 8) & (DIM - 1);
    int s = t >> 16;
    Wt[t] = Wf[n * KTOT + s * DIM + j];
}

// ---------------------------------------------------------------------------
// Kernel 2: fold  A_s[k][n] = sum_j w_s[k][j] * Wt[s][j][n]
// and store bf16 directly in WMMA B-fragment layout:
//   Bfrag[((kt_global*NNT + nt)*32 + lane)*16 + e]
//   lane = group*16 + (n%16), group = (kk>>3)&1, e = (kk&7) + (kk&16 ? 8 : 0)
// ---------------------------------------------------------------------------
__global__ void k_fold(const float* __restrict__ w_orig,
                       const float* __restrict__ w_prop,
                       const float* __restrict__ w_ctx,
                       const float* __restrict__ Wt,
                       unsigned short* __restrict__ Bfrag) {
    int t = blockIdx.x * blockDim.x + threadIdx.x;
    if (t >= 3 * DIM * DIM) return;
    int n = t & 255;
    int k = (t >> 8) & 255;
    int s = t >> 16;
    const float* w = (s == 0) ? w_orig : (s == 1) ? w_prop : w_ctx;
    const float* wrow  = w + k * DIM;                  // scalar-broadcast reads
    const float* wtcol = Wt + s * DIM * DIM + n;       // coalesced across lanes
    float acc = 0.f;
#pragma unroll 8
    for (int j = 0; j < DIM; ++j)
        acc = fmaf(wrow[j], wtcol[j * DIM], acc);

    int ktg   = s * 8 + (k >> 5);                      // global k-tile 0..23
    int kk    = k & 31;
    int group = (kk >> 3) & 1;
    int e     = (kk & 7) + ((kk & 16) ? 8 : 0);
    int lane  = group * 16 + (n & 15);
    int nt    = n >> 4;
    Bfrag[(((ktg * NNT) + nt) * 32 + lane) * 16 + e] = f32_to_bf16_rne(acc);
}

// ---------------------------------------------------------------------------
// Kernel 3: full-table copy (vectorized, grid-stride).
// Non-temporal on both sides: this is a one-pass 512MB stream; keep it from
// evicting the 384KB B-fragment buffer (and gather rows) out of the 192MB L2
// before the WMMA kernel runs.
// ---------------------------------------------------------------------------
__global__ void k_copy(const v4f* __restrict__ src,
                       v4f* __restrict__ dst, long long n4) {
    long long i      = (long long)blockIdx.x * blockDim.x + threadIdx.x;
    long long stride = (long long)gridDim.x * blockDim.x;
    for (; i < n4; i += stride) {
        v4f v = __builtin_nontemporal_load(&src[i]);
        __builtin_nontemporal_store(v, &dst[i]);
    }
}

// ---------------------------------------------------------------------------
// Kernel 4: gathered GEMM  fused = X @ B  (X: [NC,768] bf16, B folded)
// + tanh + b_fuse + bias + dual scatter store.
// Block = 256 threads (8 wave32), 32 fused rows per block.
// ---------------------------------------------------------------------------
__global__ void __launch_bounds__(256)
k_fused_gemm(const float* __restrict__ h_orig,
             const float* __restrict__ h_prop,
             const float* __restrict__ h_ctx,
             const int*   __restrict__ idx_prop,
             const int*   __restrict__ idx_ctx,
             const unsigned short* __restrict__ Bfrag,
             const float* __restrict__ b_fuse,
             const float* __restrict__ bias,
             float* __restrict__ out_prop,
             float* __restrict__ out_ctx) {
    __shared__ __align__(32) unsigned short sA[2 * NKT * 32 * 16]; // 48 KB A-fragments
    __shared__ int sIdxP[32];
    __shared__ int sIdxC[32];

    const int tid      = threadIdx.x;
    const int row_base = blockIdx.x * 32;

    if (tid < 32)       sIdxP[tid]      = idx_prop[row_base + tid];
    else if (tid < 64)  sIdxC[tid - 32] = idx_ctx[row_base + tid - 32];
    __syncthreads();

    // ---- Stage 1: gather 32 rows x 768 f32 -> bf16 A-fragments in LDS ----
    // float4 chunks: 32 * 192 = 6144 chunks, 24 per thread, coalesced reads.
    for (int c = tid; c < 32 * (KTOT / 4); c += 256) {
        int row = c / (KTOT / 4);
        int k   = (c % (KTOT / 4)) * 4;
        const float* src;
        if (k < DIM)            src = h_orig + (size_t)(row_base + row) * DIM + k;
        else if (k < 2 * DIM)   src = h_prop + (size_t)sIdxP[row] * DIM + (k - DIM);
        else                    src = h_ctx  + (size_t)sIdxC[row] * DIM + (k - 2 * DIM);
        v4f v = *(const v4f*)src;
        unsigned long long pk =
              (unsigned long long)f32_to_bf16_rne(v.x)
            | ((unsigned long long)f32_to_bf16_rne(v.y) << 16)
            | ((unsigned long long)f32_to_bf16_rne(v.z) << 32)
            | ((unsigned long long)f32_to_bf16_rne(v.w) << 48);
        // swizzle: 4 consecutive k land on 4 consecutive e in one lane slot
        int kt    = k >> 5;
        int kk    = k & 31;
        int group = (kk >> 3) & 1;
        int e0    = (kk & 7) + ((kk & 16) ? 8 : 0);
        int rt    = row >> 4;
        int lane  = group * 16 + (row & 15);
        *reinterpret_cast<unsigned long long*>(
            &sA[(((kt * 2 + rt) * 32) + lane) * 16 + e0]) = pk;
    }
    __syncthreads();

    // ---- Stage 2: 8 waves x (2 row-tiles x 2 col-tiles) of 16x16 ----
    const int wave = tid >> 5;
    const int lane = tid & 31;
    const int nt0  = wave * 2;
    const int nt1  = nt0 + 1;

    v8f acc00 = {}, acc01 = {}, acc10 = {}, acc11 = {};
    const unsigned short* bp = Bfrag + lane * 16;

#pragma unroll 4
    for (int kt = 0; kt < NKT; ++kt) {
        const v16bf a0 = *reinterpret_cast<const v16bf*>(&sA[((kt * 2 + 0) * 32 + lane) * 16]);
        const v16bf a1 = *reinterpret_cast<const v16bf*>(&sA[((kt * 2 + 1) * 32 + lane) * 16]);
        const v16bf b0 = *reinterpret_cast<const v16bf*>(bp + (size_t)((kt * NNT + nt0) * 32) * 16);
        const v16bf b1 = *reinterpret_cast<const v16bf*>(bp + (size_t)((kt * NNT + nt1) * 32) * 16);
        acc00 = __builtin_amdgcn_wmma_f32_16x16x32_bf16(false, a0, false, b0, (short)0, acc00, false, false);
        acc01 = __builtin_amdgcn_wmma_f32_16x16x32_bf16(false, a0, false, b1, (short)0, acc01, false, false);
        acc10 = __builtin_amdgcn_wmma_f32_16x16x32_bf16(false, a1, false, b0, (short)0, acc10, false, false);
        acc11 = __builtin_amdgcn_wmma_f32_16x16x32_bf16(false, a1, false, b1, (short)0, acc11, false, false);
    }

    // ---- Epilogue: tanh(acc + b_fuse) + bias, scatter to both outputs ----
    const int ncol = lane & 15;
    const int mofs = (lane >> 4) * 8;
    const int n0 = nt0 * 16 + ncol;
    const int n1 = nt1 * 16 + ncol;
    const float bf0 = b_fuse[n0], bf1 = b_fuse[n1];
    const float bi0 = bias[n0],   bi1 = bias[n1];

#pragma unroll
    for (int v = 0; v < 8; ++v) {
        int m  = v + mofs;
        int r0 = m;            // row-tile 0
        int r1 = 16 + m;       // row-tile 1
        size_t p00 = (size_t)sIdxP[r0] * DIM, c00 = (size_t)sIdxC[r0] * DIM;
        size_t p10 = (size_t)sIdxP[r1] * DIM, c10 = (size_t)sIdxC[r1] * DIM;
        float f;
        f = tanhf(acc00[v] + bf0) + bi0; out_prop[p00 + n0] = f; out_ctx[c00 + n0] = f;
        f = tanhf(acc01[v] + bf1) + bi1; out_prop[p00 + n1] = f; out_ctx[c00 + n1] = f;
        f = tanhf(acc10[v] + bf0) + bi0; out_prop[p10 + n0] = f; out_ctx[c10 + n0] = f;
        f = tanhf(acc11[v] + bf1) + bi1; out_prop[p10 + n1] = f; out_ctx[c10 + n1] = f;
    }
}

// ---------------------------------------------------------------------------
extern "C" void kernel_launch(void* const* d_in, const int* in_sizes, int n_in,
                              void* d_out, int out_size, void* d_ws, size_t ws_size,
                              hipStream_t stream) {
    (void)in_sizes; (void)n_in; (void)out_size; (void)ws_size;
    const float* h_prop  = (const float*)d_in[0];
    const float* h_ctx   = (const float*)d_in[1];
    const float* h_orig  = (const float*)d_in[2];
    const float* w_orig  = (const float*)d_in[3];
    const float* w_prop  = (const float*)d_in[4];
    const float* w_ctx   = (const float*)d_in[5];
    const float* W_fuse  = (const float*)d_in[6];
    const float* b_fuse  = (const float*)d_in[7];
    const float* bias    = (const float*)d_in[8];
    const int*   idx_p   = (const int*)d_in[9];
    const int*   idx_c   = (const int*)d_in[10];

    float* out_prop = (float*)d_out;
    float* out_ctx  = out_prop + (size_t)N_NODES * DIM;

    // workspace: [0, 384KB) bf16 B fragments; [512KB, 512KB+768KB) f32 Wt
    unsigned short* Bfrag = (unsigned short*)d_ws;
    float*          Wt    = (float*)((char*)d_ws + 512 * 1024);

    // 1) transpose W_fuse (196608 elems)
    k_transpose_wfuse<<<768, 256, 0, stream>>>(W_fuse, Wt);
    // 2) fold w_s @ Wf_s^T into bf16 B fragments
    k_fold<<<768, 256, 0, stream>>>(w_orig, w_prop, w_ctx, Wt, Bfrag);
    // 3) full-table copies into both outputs (non-temporal streaming)
    long long n4 = (long long)N_NODES * DIM / 4;
    k_copy<<<4096, 256, 0, stream>>>((const v4f*)h_prop, (v4f*)out_prop, n4);
    k_copy<<<4096, 256, 0, stream>>>((const v4f*)h_ctx,  (v4f*)out_ctx,  n4);
    // 4) fused gathered-GEMM + activation + dual scatter
    k_fused_gemm<<<NC_NODES / 32, 256, 0, stream>>>(
        h_orig, h_prop, h_ctx, idx_p, idx_c, Bfrag, b_fuse, bias, out_prop, out_ctx);
}
